// GRUModel_2877628088777
// MI455X (gfx1250) — compile-verified, matrix operand order checked by reading, not run on hardware
//
#include <hip/hip_runtime.h>
#include <hip/hip_bf16.h>
#include <math.h>
#include <stdint.h>

// ---------------- problem constants ----------------
#define VOCAB 32000
#define DIN   512
#define DH    1024
#define G3    3072      // 3*DH
#define DOUT  64
#define BATCH 64
#define SEQ   256
#define MROWS (BATCH * SEQ)   // 16384
#define NBLK_STEP 64          // persistent-kernel workgroups (DH/16)
#define LDSW (DH + 8)         // padded LDS row stride (elements) to break bank pattern

typedef __attribute__((ext_vector_type(16))) __bf16 v16bf;
typedef __attribute__((ext_vector_type(8)))  float  v8f;

// ---------------- helpers ----------------
__device__ __forceinline__ unsigned short f32_to_bf16(float f) {
  unsigned int u = __float_as_uint(f);
  unsigned int r = (u + 0x7FFFu + ((u >> 16) & 1u)) >> 16;  // round-to-nearest-even
  return (unsigned short)r;
}

union Frag16 { uint4 q[2]; v16bf v; };

// Load a 16x32 bf16 WMMA fragment from row-major memory (stride in elements).
// ISA layout: lane m=lane&15 -> matrix row; VGPR0-3 = K[kh*8..kh*8+7],
// VGPR4-7 = K[16+kh*8..16+kh*8+7] (kh=lane>>4). Two 16B loads per lane.
__device__ __forceinline__ v16bf load_frag(const unsigned short* __restrict__ base,
                                           int stride) {
  int lane = threadIdx.x & 31;
  const unsigned short* p = base + (long long)(lane & 15) * stride + ((lane >> 4) << 3);
  Frag16 f;
  f.q[0] = *(const uint4*)(p);
  f.q[1] = *(const uint4*)(p + 16);
  return f.v;
}

__device__ __forceinline__ v8f wmma_bf16(v16bf a, v16bf b, v8f c) {
  return __builtin_amdgcn_wmma_f32_16x16x32_bf16(false, a, false, b, (short)0, c,
                                                 false, false);
}

// ---------------- kernel 1: f32 -> bf16 weight conversion ----------------
__global__ void convert_bf16_kernel(const float* __restrict__ in,
                                    unsigned short* __restrict__ out, int n) {
  int i = blockIdx.x * blockDim.x + threadIdx.x;
  if (i < n) out[i] = f32_to_bf16(in[i]);
}

// ---------------- kernel 2: embedding gather -> bf16, permuted to [s][b][d] ----------------
__global__ void embed_gather_kernel(const int* __restrict__ x,
                                    const float* __restrict__ emb,
                                    unsigned short* __restrict__ xe) {
  int bs = blockIdx.x;              // b*SEQ + s  (x is [B][S])
  int b = bs / SEQ, s = bs % SEQ;
  int tok = x[bs];
  const float* src = emb + (long long)tok * DIN;
  unsigned short* dst = xe + (long long)(s * BATCH + b) * DIN;  // row m = s*64+b
  for (int d = threadIdx.x; d < DIN; d += blockDim.x) dst[d] = f32_to_bf16(src[d]);
}

// ---------------- kernel 3: zero hidden state + step-barrier counters ----------------
__global__ void init_state_kernel(unsigned short* __restrict__ hbf0,
                                  unsigned short* __restrict__ hbf1,
                                  float* __restrict__ hf0, float* __restrict__ hf1,
                                  unsigned int* __restrict__ bar) {
  int i = blockIdx.x * blockDim.x + threadIdx.x;
  if (i < BATCH * DH) { hbf0[i] = 0; hbf1[i] = 0; hf0[i] = 0.f; hf1[i] = 0.f; }
  if (i < SEQ) bar[i] = 0u;
}

// ---------------- kernel 4: gi = xe @ w_ih^T + b_ih  (M=16384,K=512,N=3072) ----------------
// 256 threads = 8 waves: 4 along M x 2 along N. Block tile 64(M) x 128(N).
__global__ void __launch_bounds__(256)
gemm_gi_kernel(const unsigned short* __restrict__ xe,
               const unsigned short* __restrict__ wih,
               const float* __restrict__ b_ih,
               float* __restrict__ gi) {
  int wave = threadIdx.x >> 5;
  int lane = threadIdx.x & 31;
  int m0 = blockIdx.y * 64 + (wave >> 1) * 16;
  int n0 = blockIdx.x * 128 + (wave & 1) * 64;

  v8f acc[4] = {};
  for (int k = 0; k < DIN; k += 32) {
    v16bf a = load_frag(xe + (long long)m0 * DIN + k, DIN);
#pragma unroll
    for (int t = 0; t < 4; ++t) {
      v16bf b = load_frag(wih + (long long)(n0 + t * 16) * DIN + k, DIN);
      acc[t] = wmma_bf16(a, b, acc[t]);
    }
  }
  int ml = (lane >> 4) << 3;
  int nl = lane & 15;
#pragma unroll
  for (int t = 0; t < 4; ++t) {
    int n = n0 + t * 16 + nl;
    float bias = b_ih[n];
#pragma unroll
    for (int v = 0; v < 8; ++v) {
      int m = m0 + ml + v;                      // m = s*64 + b
      gi[(long long)m * G3 + n] = acc[t][v] + bias;
    }
  }
}

// ---------------- kernel 5: persistent GRU recurrence ----------------
// 64 workgroups x 128 threads. Each block owns 16 hidden units; stages its
// w_hh slice (3 gates x 16 rows x 1024, bf16) into LDS once, then iterates all
// 256 timesteps with a grid-wide barrier per step (counters pre-zeroed).
__global__ void __launch_bounds__(128)
gru_persistent_kernel(const unsigned short* __restrict__ whh,
                      const float* __restrict__ b_hh,
                      const float* __restrict__ gi,
                      unsigned short* __restrict__ h_bf0,
                      float* __restrict__ h_f0,
                      unsigned short* __restrict__ h_bf1,
                      float* __restrict__ h_f1,
                      unsigned short* __restrict__ hs,
                      unsigned int* __restrict__ bar) {
  extern __shared__ unsigned short lds_w[];   // [48][LDSW]

  const int wave = threadIdx.x >> 5;
  const int lane = threadIdx.x & 31;
  const int m0 = wave * 16;                   // batch rows for this wave
  const int j0 = blockIdx.x * 16;             // hidden-unit slice for this block

  // ---- stage w_hh slice into LDS (rows: g*16+rr -> global row g*DH+j0+rr) ----
  for (int idx = threadIdx.x; idx < 48 * (DH / 8); idx += blockDim.x) {
    int r = idx / (DH / 8);
    int c = idx % (DH / 8);
    int g = r >> 4, rr = r & 15;
    uint4 val = *((const uint4*)(whh + (long long)(g * DH + j0 + rr) * DH) + c);
    *(uint4*)(lds_w + r * LDSW + c * 8) = val;
  }
  __syncthreads();

  const int jl = j0 + (lane & 15);
  const float bh_r = b_hh[jl];
  const float bh_z = b_hh[DH + jl];
  const float bh_n = b_hh[2 * DH + jl];
  const int ml = (lane >> 4) << 3;

  for (int s = 0; s < SEQ; ++s) {
    const unsigned short* hin_bf = (s & 1) ? h_bf1 : h_bf0;
    const float*          hin_f  = (s & 1) ? h_f1 : h_f0;
    unsigned short*       hout_bf = (s & 1) ? h_bf0 : h_bf1;
    float*                hout_f  = (s & 1) ? h_f0 : h_f1;

    v8f cr = {}, cz = {}, cn = {};
    for (int k = 0; k < DH; k += 32) {
      v16bf a  = load_frag(hin_bf + (long long)m0 * DH + k, DH);
      v16bf br = load_frag(lds_w + 0 * 16 * LDSW + k, LDSW);
      v16bf bz = load_frag(lds_w + 1 * 16 * LDSW + k, LDSW);
      v16bf bn = load_frag(lds_w + 2 * 16 * LDSW + k, LDSW);
      cr = wmma_bf16(a, br, cr);
      cz = wmma_bf16(a, bz, cz);
      cn = wmma_bf16(a, bn, cn);
    }

    const float* gib = gi + (long long)s * BATCH * G3;
#pragma unroll
    for (int v = 0; v < 8; ++v) {
      int b = m0 + ml + v;
      const float* g = gib + (long long)b * G3;
      float r = 1.f / (1.f + __expf(-(g[jl]          + cr[v] + bh_r)));
      float z = 1.f / (1.f + __expf(-(g[DH + jl]     + cz[v] + bh_z)));
      float n = tanhf(g[2 * DH + jl] + r * (cn[v] + bh_n));
      float hn = (1.f - z) * n + z * hin_f[(long long)b * DH + jl];
      hout_f[(long long)b * DH + jl] = hn;
      unsigned short hb = f32_to_bf16(hn);
      hout_bf[(long long)b * DH + jl] = hb;
      hs[((long long)b * SEQ + s) * DH + jl] = hb;   // FC row order m = b*S + s
    }

    // ---- grid-wide barrier for step s (release add, acquire spin) ----
    __syncthreads();
    if (threadIdx.x == 0) {
      __threadfence();
      atomicAdd(&bar[s], 1u);
      while (__hip_atomic_load(&bar[s], __ATOMIC_ACQUIRE,
                               __HIP_MEMORY_SCOPE_AGENT) < (unsigned)NBLK_STEP) {
        __builtin_amdgcn_s_sleep(1);
      }
    }
    __syncthreads();
  }
}

// ---------------- kernel 6: logits = hs @ fc_w^T + fc_b, then log_softmax over 64 ----------------
__global__ void __launch_bounds__(128)
fc_logsoftmax_kernel(const unsigned short* __restrict__ hs,
                     const unsigned short* __restrict__ fcw,
                     const float* __restrict__ fcb,
                     float* __restrict__ out) {
  __shared__ float tile[4][16][DOUT + 1];
  int wave = threadIdx.x >> 5;
  int lane = threadIdx.x & 31;
  int m0 = blockIdx.x * 64 + wave * 16;

  v8f acc[4] = {};
  for (int k = 0; k < DH; k += 32) {
    v16bf a = load_frag(hs + (long long)m0 * DH + k, DH);
#pragma unroll
    for (int t = 0; t < 4; ++t) {
      v16bf b = load_frag(fcw + (long long)(t * 16) * DH + k, DH);
      acc[t] = wmma_bf16(a, b, acc[t]);
    }
  }
  int ml = (lane >> 4) << 3;
  int nl = lane & 15;
#pragma unroll
  for (int t = 0; t < 4; ++t) {
    float bias = fcb[t * 16 + nl];
#pragma unroll
    for (int v = 0; v < 8; ++v) tile[wave][ml + v][t * 16 + nl] = acc[t][v] + bias;
  }
  __syncthreads();
  if (lane < 16) {
    float* row = tile[wave][lane];
    float mx = row[0];
    for (int o = 1; o < DOUT; ++o) mx = fmaxf(mx, row[o]);
    float sum = 0.f;
    for (int o = 0; o < DOUT; ++o) sum += __expf(row[o] - mx);
    float lse = mx + __logf(sum);
    float* dst = out + (long long)(m0 + lane) * DOUT;
    for (int o = 0; o < DOUT; ++o) dst[o] = row[o] - lse;
  }
}

// ---------------- host orchestration ----------------
extern "C" void kernel_launch(void* const* d_in, const int* in_sizes, int n_in,
                              void* d_out, int out_size, void* d_ws, size_t ws_size,
                              hipStream_t stream) {
  const int*   x    = (const int*)d_in[0];
  const float* emb  = (const float*)d_in[1];
  const float* w_ih = (const float*)d_in[2];
  const float* w_hh = (const float*)d_in[3];
  const float* b_ih = (const float*)d_in[4];
  const float* b_hh = (const float*)d_in[5];
  const float* fc_w = (const float*)d_in[6];
  const float* fc_b = (const float*)d_in[7];
  float* out = (float*)d_out;

  // workspace carve-up (256B aligned)
  char* w = (char*)d_ws;
  auto alloc = [&](size_t bytes) {
    char* p = w;
    w += (bytes + 255) & ~(size_t)255;
    return p;
  };
  unsigned short* wih_bf = (unsigned short*)alloc((size_t)G3 * DIN * 2);
  unsigned short* whh_bf = (unsigned short*)alloc((size_t)G3 * DH * 2);
  unsigned short* fcw_bf = (unsigned short*)alloc((size_t)DOUT * DH * 2);
  unsigned short* xe_bf  = (unsigned short*)alloc((size_t)MROWS * DIN * 2);
  unsigned short* hs_bf  = (unsigned short*)alloc((size_t)MROWS * DH * 2);
  unsigned short* h_bf0  = (unsigned short*)alloc((size_t)BATCH * DH * 2);
  unsigned short* h_bf1  = (unsigned short*)alloc((size_t)BATCH * DH * 2);
  float*          h_f0   = (float*)alloc((size_t)BATCH * DH * 4);
  float*          h_f1   = (float*)alloc((size_t)BATCH * DH * 4);
  unsigned int*   bar    = (unsigned int*)alloc((size_t)SEQ * 4);
  float*          gi     = (float*)alloc((size_t)MROWS * G3 * 4);

  // 1) weights -> bf16
  {
    int n = G3 * DIN;
    convert_bf16_kernel<<<(n + 255) / 256, 256, 0, stream>>>(w_ih, wih_bf, n);
  }
  {
    int n = G3 * DH;
    convert_bf16_kernel<<<(n + 255) / 256, 256, 0, stream>>>(w_hh, whh_bf, n);
  }
  {
    int n = DOUT * DH;
    convert_bf16_kernel<<<(n + 255) / 256, 256, 0, stream>>>(fc_w, fcw_bf, n);
  }

  // 2) embedding gather (permuted to [s][b][d]) + 3) init h0/barriers
  embed_gather_kernel<<<BATCH * SEQ, 256, 0, stream>>>(x, emb, xe_bf);
  init_state_kernel<<<(BATCH * DH + 255) / 256, 256, 0, stream>>>(h_bf0, h_bf1, h_f0,
                                                                  h_f1, bar);

  // 4) gi = xe @ w_ih^T + b_ih   [rows ordered s*64+b -> scan order]
  {
    dim3 grid(G3 / 128, MROWS / 64);
    gemm_gi_kernel<<<grid, 256, 0, stream>>>(xe_bf, wih_bf, b_ih, gi);
  }

  // 5) persistent recurrence: one kernel, 256 internal steps, grid barriers.
  //    Dynamic LDS: 48 rows x (DH+8) bf16 elements.
  {
    size_t lds_bytes = (size_t)48 * LDSW * sizeof(unsigned short);
    gru_persistent_kernel<<<NBLK_STEP, 128, lds_bytes, stream>>>(
        whh_bf, b_hh, gi, h_bf0, h_f0, h_bf1, h_f1, hs_bf, bar);
  }

  // 6) FC + log_softmax
  fc_logsoftmax_kernel<<<MROWS / 64, 128, 0, stream>>>(hs_bf, fcw_bf, fc_b, out);

  (void)in_sizes; (void)n_in; (void)out_size; (void)ws_size;
}
